// MamboPoolingWithNodeAttention_43104291783364
// MI455X (gfx1250) — matrix operand order repaired and checked by reading, not compiled, not run on hardware
//
#include <hip/hip_runtime.h>
#include <hip/hip_bf16.h>

typedef __attribute__((ext_vector_type(16))) _Float16 v16h;
typedef __attribute__((ext_vector_type(8)))  float    v8f;

#define N_NODES 50000
#define C_DIM   256
#define E_EDGES 800000
#define N_TILES (N_NODES / 16)          /* 3125, exact */
#define NEG_SLOPE 0.2f
#define SORT_BLK 1024
#define NB_SORT ((E_EDGES + SORT_BLK - 1) / SORT_BLK)
#define ENC_NEG_INF 0x007FFFFFu

// ---- order-preserving float <-> uint encoding for atomic max ----
__device__ __forceinline__ unsigned fenc(float f) {
    unsigned u = __float_as_uint(f);
    return (u & 0x80000000u) ? ~u : (u | 0x80000000u);
}
__device__ __forceinline__ float fdec(unsigned u) {
    unsigned v = (u & 0x80000000u) ? (u & 0x7FFFFFFFu) : ~u;
    return __uint_as_float(v);
}

// ============================================================
// K1: fold attention vectors through W:  w_src[k] = sum_j att_src[j]*W[j,k]
// (collapses the 6.5 GFLOP h-GEMM into a 256-wide vector)
// ============================================================
__global__ void fold_att(const float* __restrict__ W,
                         const float* __restrict__ att_src,
                         const float* __restrict__ att_dst,
                         float* __restrict__ w_src, float* __restrict__ w_dst) {
    int k = threadIdx.x; // 256 threads
    float ss = 0.f, sd = 0.f;
    for (int j = 0; j < C_DIM; ++j) {
        float w = W[j * C_DIM + k];
        ss += att_src[j] * w;
        sd += att_dst[j] * w;
    }
    w_src[k] = ss;
    w_dst[k] = sd;
}

// ============================================================
// K2: fused dual GEMV via WMMA. One wave handles TWO 16-row tiles of x.
// Full 16x256 tiles staged to LDS as f16 once; K-loop fully unrolled
// (16 static v_wmma). bvec (cols 0/1 = w_src/w_dst) shared across tiles.
// ============================================================
__global__ void __launch_bounds__(32)
dual_gemv_wmma(const float* __restrict__ x,
               const float* __restrict__ w_src,
               const float* __restrict__ w_dst,
               float* __restrict__ a_src, float* __restrict__ a_dst) {
    __shared__ _Float16 Atile[2][16 * C_DIM];   // 16 KB
    const int lane = threadIdx.x;               // wave32, full EXEC throughout
    const int t0   = blockIdx.x * 2;
    const bool has2 = (t0 + 1) < N_TILES;
    const int t1   = has2 ? (t0 + 1) : t0;      // clamp: duplicate compute, guarded store

    // prefetch next block's first tile (global_prefetch_b8)
    if (t0 + 2 < N_TILES)
        __builtin_prefetch(x + (size_t)(t0 + 2) * 16 * C_DIM + lane * 16, 0, 1);

    // ---- stage both 16x256 f32 tiles into LDS as f16 (coalesced b128 pairs)
    #pragma unroll
    for (int tt = 0; tt < 2; ++tt) {
        const float* base = x + (size_t)(tt ? t1 : t0) * 16 * C_DIM;
        #pragma unroll
        for (int r = 0; r < 16; ++r) {
            const float* p = base + r * C_DIM + lane * 8;
            float4 f0 = *(const float4*)p;
            float4 f1 = *(const float4*)(p + 4);
            _Float16* q = &Atile[tt][r * C_DIM + lane * 8];
            q[0] = (_Float16)f0.x; q[1] = (_Float16)f0.y;
            q[2] = (_Float16)f0.z; q[3] = (_Float16)f0.w;
            q[4] = (_Float16)f1.x; q[5] = (_Float16)f1.y;
            q[6] = (_Float16)f1.z; q[7] = (_Float16)f1.w;
        }
    }
    __syncthreads(); // single-wave WG -> S_NOP

    const int n_col = lane & 15;            // B/D column this lane owns
    const int kb    = (lane >> 4) * 16;     // B K-half base (0 or 16)
    const int m_row = lane & 15;            // A row this lane owns
    const int khalf = lane >> 4;            // A K-half selector
    const float* wp = (n_col == 1) ? w_dst : w_src;
    const bool w_active = (n_col < 2);

    v8f acc0 = {}, acc1 = {};
    #pragma unroll
    for (int c = 0; c < 8; ++c) {
        const int k0 = c * 32;
        // B: 32x16, lane holds column n_col; only cols 0/1 load (exec-masked)
        v16h bvec;
        #pragma unroll
        for (int e = 0; e < 16; ++e) {
            float v = w_active ? wp[k0 + kb + e] : 0.0f;
            bvec[e] = (_Float16)v;
        }
        // A: contiguous K layout matching B: lanes<16: K=e ; lanes>=16: K=16+e
        v16h avec0, avec1;
        #pragma unroll
        for (int e = 0; e < 16; ++e) {
            int K = k0 + khalf * 16 + e;
            avec0[e] = Atile[0][m_row * C_DIM + K];
            avec1[e] = Atile[1][m_row * C_DIM + K];
        }
        acc0 = __builtin_amdgcn_wmma_f32_16x16x32_f16(
            false, avec0, false, bvec, (short)0, acc0, false, false);
        acc1 = __builtin_amdgcn_wmma_f32_16x16x32_f16(
            false, avec1, false, bvec, (short)0, acc1, false, false);
    }

    // D layout: element v -> row M = v + (lane>=16 ? 8 : 0), col = lane&15
    const int roff = (lane >= 16) ? 8 : 0;
    if (n_col == 0) {
        #pragma unroll
        for (int v = 0; v < 8; ++v) a_src[t0 * 16 + roff + v] = acc0[v];
        if (has2) {
            #pragma unroll
            for (int v = 0; v < 8; ++v) a_src[t1 * 16 + roff + v] = acc1[v];
        }
    } else if (n_col == 1) {
        #pragma unroll
        for (int v = 0; v < 8; ++v) a_dst[t0 * 16 + roff + v] = acc0[v];
        if (has2) {
            #pragma unroll
            for (int v = 0; v < 8; ++v) a_dst[t1 * 16 + roff + v] = acc1[v];
        }
    }
}

// ============================================================
// K3: per-node init
// ============================================================
__global__ void init_nodes(unsigned* m1, float* s1, unsigned* m2, float* s2,
                           int* mask, float* out_score, int* out_batch) {
    int i = blockIdx.x * blockDim.x + threadIdx.x;
    if (i >= N_NODES) return;
    m1[i] = ENC_NEG_INF; s1[i] = 0.f;
    m2[i] = ENC_NEG_INF; s2[i] = 0.f;
    mask[i] = 1;
    out_score[i] = 1.0f;
    out_batch[i] = 0;
}

__global__ void zero_f32(float* p, long n) {
    long i = (long)blockIdx.x * blockDim.x + threadIdx.x;
    if (i < n) p[i] = 0.f;
}

// ============================================================
// K4..K8: segment softmaxes (E+N with self loops, then E)
// ============================================================
__global__ void edge_raw_max(const int* __restrict__ src, const int* __restrict__ dst,
                             const float* __restrict__ a_src, const float* __restrict__ a_dst,
                             float* __restrict__ raw, unsigned* __restrict__ m1) {
    int i = blockIdx.x * blockDim.x + threadIdx.x;
    if (i >= E_EDGES + N_NODES) return;
    int s, d;
    if (i < E_EDGES) { s = src[i]; d = dst[i]; } else { s = d = i - E_EDGES; }
    float r = a_src[s] + a_dst[d];
    r = (r >= 0.f) ? r : NEG_SLOPE * r;
    raw[i] = r;
    atomicMax(&m1[d], fenc(r));
}

__global__ void edge_exp_sum(const int* __restrict__ dst,
                             float* __restrict__ raw, const unsigned* __restrict__ m1,
                             float* __restrict__ s1) {
    int i = blockIdx.x * blockDim.x + threadIdx.x;
    if (i >= E_EDGES + N_NODES) return;
    int d = (i < E_EDGES) ? dst[i] : (i - E_EDGES);
    float m = fdec(m1[d]);
    if (m == -__builtin_inff()) m = 0.f;
    float ex = __expf(raw[i] - m);
    raw[i] = ex;                       // overwrite with numerator
    atomicAdd(&s1[d], ex);
}

__global__ void alpha_max2(const int* __restrict__ dst,
                           const float* __restrict__ exbuf, const float* __restrict__ s1,
                           float* __restrict__ ealpha, unsigned* __restrict__ m2) {
    int i = blockIdx.x * blockDim.x + threadIdx.x;
    if (i >= E_EDGES) return;
    int d = dst[i];
    float a = exbuf[i] / (s1[d] + 1e-16f);
    ealpha[i] = a;
    atomicMax(&m2[d], fenc(a));
}

__global__ void exp2_sum2(const int* __restrict__ dst,
                          const float* __restrict__ ealpha, const unsigned* __restrict__ m2,
                          float* __restrict__ ex2, float* __restrict__ s2) {
    int i = blockIdx.x * blockDim.x + threadIdx.x;
    if (i >= E_EDGES) return;
    int d = dst[i];
    float m = fdec(m2[d]);
    if (m == -__builtin_inff()) m = 0.f;
    float ex = __expf(ealpha[i] - m);
    ex2[i] = ex;
    atomicAdd(&s2[d], ex);
}

__global__ void score_key(const int* __restrict__ dst,
                          const float* __restrict__ ex2, const float* __restrict__ s2,
                          float* __restrict__ score, unsigned* __restrict__ keys,
                          int* __restrict__ idx) {
    int i = blockIdx.x * blockDim.x + threadIdx.x;
    if (i >= E_EDGES) return;
    int d = dst[i];
    float sc = ex2[i] / (s2[d] + 1e-16f);
    score[i] = sc;
    keys[i] = ~fenc(sc);   // ascending radix sort => descending score
    idx[i] = i;
}

// ============================================================
// K9..K11: stable LSD radix sort (4 passes, 8-bit digits)
// ============================================================
__global__ void radix_hist(const unsigned* __restrict__ keys, int n, int shift,
                           unsigned* __restrict__ counts) {
    __shared__ unsigned hist[256];
    hist[threadIdx.x] = 0;
    __syncthreads();
    int base = blockIdx.x * SORT_BLK;
    for (int j = threadIdx.x; j < SORT_BLK; j += 256) {
        int i = base + j;
        if (i < n) atomicAdd(&hist[(keys[i] >> shift) & 255u], 1u);
    }
    __syncthreads();
    counts[threadIdx.x * NB_SORT + blockIdx.x] = hist[threadIdx.x];
}

__global__ void radix_scan(unsigned* __restrict__ counts) {
    __shared__ unsigned tot[256];
    unsigned sum = 0;
    for (int b = 0; b < NB_SORT; ++b) sum += counts[threadIdx.x * NB_SORT + b];
    tot[threadIdx.x] = sum;
    __syncthreads();
    if (threadIdx.x == 0) {
        unsigned run = 0;
        for (int d = 0; d < 256; ++d) { unsigned t = tot[d]; tot[d] = run; run += t; }
    }
    __syncthreads();
    unsigned run = tot[threadIdx.x];
    for (int b = 0; b < NB_SORT; ++b) {
        unsigned t = counts[threadIdx.x * NB_SORT + b];
        counts[threadIdx.x * NB_SORT + b] = run;
        run += t;
    }
}

__global__ void radix_scatter(const unsigned* __restrict__ keysIn, const int* __restrict__ idxIn,
                              unsigned* __restrict__ keysOut, int* __restrict__ idxOut,
                              const unsigned* __restrict__ counts, int n, int shift) {
    __shared__ unsigned off[256];
    off[threadIdx.x] = counts[threadIdx.x * NB_SORT + blockIdx.x];
    __syncthreads();
    if (threadIdx.x == 0) {        // serial within block => stable
        int base = blockIdx.x * SORT_BLK;
        int end = base + SORT_BLK; if (end > n) end = n;
        for (int i = base; i < end; ++i) {
            unsigned k = keysIn[i];
            unsigned d = (k >> shift) & 255u;
            unsigned p = off[d]++;
            keysOut[p] = k;
            idxOut[p] = idxIn[i];
        }
    }
}

// ============================================================
// K12: greedy matching scan (inherently sequential, mirrors lax.scan)
// ============================================================
__global__ void greedy_scan(const int* __restrict__ idxSorted,
                            const int* __restrict__ src, const int* __restrict__ dst,
                            const float* __restrict__ score,
                            int* __restrict__ mask, int* __restrict__ cluster,
                            float* __restrict__ out_score) {
    if (threadIdx.x != 0) return;
    int cnt = 0;
    for (int t = 0; t < E_EDGES; ++t) {
        int id = idxSorted[t];
        int s = src[id], d = dst[id];
        if (mask[s] && mask[d]) {
            cluster[s] = cnt; cluster[d] = cnt;
            out_score[cnt] = score[id];
            mask[s] = 0; mask[d] = 0;
            ++cnt;
        }
    }
    int run = cnt;
    for (int i = 0; i < N_NODES; ++i)
        if (mask[i]) cluster[i] = run++;
}

// ============================================================
// K13..K16: outputs
// ============================================================
__global__ void scatter_x(const float* __restrict__ x, const int* __restrict__ cluster,
                          float* __restrict__ out_x) {
    long t = (long)blockIdx.x * blockDim.x + threadIdx.x;
    if (t >= (long)N_NODES * C_DIM) return;
    int i = (int)(t >> 8);            // C == 256
    int c = (int)(t & 255);
    atomicAdd(&out_x[(size_t)cluster[i] * C_DIM + c], x[t]);
}

__global__ void scale_x(float* __restrict__ out_x, const float* __restrict__ out_score) {
    long t = (long)blockIdx.x * blockDim.x + threadIdx.x;
    if (t >= (long)N_NODES * C_DIM) return;
    out_x[t] *= out_score[t >> 8];
}

__global__ void remap_edges(const int* __restrict__ ei, const int* __restrict__ cluster,
                            int* __restrict__ out_ei) {
    int j = blockIdx.x * blockDim.x + threadIdx.x;
    if (j >= 2 * E_EDGES) return;
    out_ei[j] = cluster[ei[j]];
}

__global__ void scatter_batch(const int* __restrict__ batch, const int* __restrict__ cluster,
                              int* __restrict__ out_batch) {
    int i = blockIdx.x * blockDim.x + threadIdx.x;
    if (i >= N_NODES) return;
    out_batch[cluster[i]] = batch[i];
}

// ============================================================
extern "C" void kernel_launch(void* const* d_in, const int* in_sizes, int n_in,
                              void* d_out, int out_size, void* d_ws, size_t ws_size,
                              hipStream_t stream) {
    const float* x        = (const float*)d_in[0];
    const int*   ei       = (const int*)d_in[1];
    const int*   batch    = (const int*)d_in[2];
    const float* W        = (const float*)d_in[3];
    const float* att_src  = (const float*)d_in[4];
    const float* att_dst  = (const float*)d_in[5];
    const int* src = ei;
    const int* dst = ei + E_EDGES;

    // ---- output layout: new_x | new_edge_index | new_batch | cluster | new_edge_score
    float* out_x       = (float*)d_out;
    int*   out_ei      = (int*)(out_x + (size_t)N_NODES * C_DIM);
    int*   out_batch   = out_ei + 2 * E_EDGES;
    int*   out_cluster = out_batch + N_NODES;
    float* out_score   = (float*)(out_cluster + N_NODES);

    // ---- workspace partition (4-byte words)
    float* w = (float*)d_ws;
    float*    w_src  = w;                       w += C_DIM;
    float*    w_dst  = w;                       w += C_DIM;
    float*    a_src  = w;                       w += N_NODES;
    float*    a_dst  = w;                       w += N_NODES;
    unsigned* m1     = (unsigned*)w;            w += N_NODES;
    float*    s1     = w;                       w += N_NODES;
    unsigned* m2     = (unsigned*)w;            w += N_NODES;
    float*    s2     = w;                       w += N_NODES;
    float*    rawbuf = w;                       w += (E_EDGES + N_NODES);
    float*    ealpha = w;                       w += E_EDGES;
    float*    ex2    = w;                       w += E_EDGES;
    float*    score  = w;                       w += E_EDGES;
    unsigned* keysA  = (unsigned*)w;            w += E_EDGES;
    int*      idxA   = (int*)w;                 w += E_EDGES;
    unsigned* keysB  = (unsigned*)w;            w += E_EDGES;
    int*      idxB   = (int*)w;                 w += E_EDGES;
    unsigned* counts = (unsigned*)w;            w += 256 * NB_SORT;
    int*      mask   = (int*)w;                 w += N_NODES;

    const int TB = 256;
    const long NC = (long)N_NODES * C_DIM;

    // 1) fold att through W; 2) WMMA dual GEMV for a_src/a_dst
    fold_att<<<1, 256, 0, stream>>>(W, att_src, att_dst, w_src, w_dst);
    dual_gemv_wmma<<<(N_TILES + 1) / 2, 32, 0, stream>>>(x, w_src, w_dst, a_src, a_dst);

    // 3) init
    init_nodes<<<(N_NODES + TB - 1) / TB, TB, 0, stream>>>(m1, s1, m2, s2, mask, out_score, out_batch);
    zero_f32<<<(int)((NC + TB - 1) / TB), TB, 0, stream>>>(out_x, NC);

    // 4) first segment softmax over E+N (with self loops)
    int gEN = (E_EDGES + N_NODES + TB - 1) / TB;
    edge_raw_max<<<gEN, TB, 0, stream>>>(src, dst, a_src, a_dst, rawbuf, m1);
    edge_exp_sum<<<gEN, TB, 0, stream>>>(dst, rawbuf, m1, s1);

    // 5) second segment softmax over real edges
    int gE = (E_EDGES + TB - 1) / TB;
    alpha_max2<<<gE, TB, 0, stream>>>(dst, rawbuf, s1, ealpha, m2);
    exp2_sum2<<<gE, TB, 0, stream>>>(dst, ealpha, m2, ex2, s2);
    score_key<<<gE, TB, 0, stream>>>(dst, ex2, s2, score, keysA, idxA);

    // 6) stable LSD radix sort, 4 passes, ping-pong A<->B (ends in A)
    unsigned* ki = keysA; int* ii = idxA;
    unsigned* ko = keysB; int* io = idxB;
    for (int pass = 0; pass < 4; ++pass) {
        int shift = pass * 8;
        radix_hist<<<NB_SORT, 256, 0, stream>>>(ki, E_EDGES, shift, counts);
        radix_scan<<<1, 256, 0, stream>>>(counts);
        radix_scatter<<<NB_SORT, 256, 0, stream>>>(ki, ii, ko, io, counts, E_EDGES, shift);
        unsigned* tk = ki; ki = ko; ko = tk;
        int* ti = ii; ii = io; io = ti;
    }
    // after even #passes, sorted (key,idx) live in keysA/idxA (== ki/ii)

    // 7) sequential greedy matching
    greedy_scan<<<1, 32, 0, stream>>>(ii, src, dst, score, mask, out_cluster, out_score);

    // 8) outputs
    scatter_x<<<(int)((NC + TB - 1) / TB), TB, 0, stream>>>(x, out_cluster, out_x);
    scale_x<<<(int)((NC + TB - 1) / TB), TB, 0, stream>>>(out_x, out_score);
    remap_edges<<<(2 * E_EDGES + TB - 1) / TB, TB, 0, stream>>>(ei, out_cluster, out_ei);
    scatter_batch<<<(N_NODES + TB - 1) / TB, TB, 0, stream>>>(batch, out_cluster, out_batch);
}
